// HGNN_conv_86895778333083
// MI455X (gfx1250) — matrix operand order repaired and checked by reading, not compiled, not run on hardware
//
#include <hip/hip_runtime.h>
#include <hip/hip_bf16.h>

typedef __attribute__((ext_vector_type(16))) __bf16         v16bf;
typedef __attribute__((ext_vector_type(8)))  float          v8f;
typedef __attribute__((ext_vector_type(8)))  unsigned short u16x8;
typedef __attribute__((ext_vector_type(4)))  unsigned short u16x4;
typedef __attribute__((ext_vector_type(4)))  float          f32x4;
typedef int v4i __attribute__((vector_size(16)));           // matches builtin proto

#define NN 8192   // nodes
#define NE 4096   // edges
#define FT 512    // in/out features

#define TM 128
#define TN 128
#define TK 32
#define KPAD 40   // padded LDS K-stride (80B = 20 banks, keeps 16B alignment)

#ifndef __has_builtin
#define __has_builtin(x) 0
#endif
#if __has_builtin(__builtin_amdgcn_global_load_async_to_lds_b128)
#define ASYNC_LDS 1
#else
#define ASYNC_LDS 0
#endif

typedef __attribute__((address_space(1))) v4i gv4i_t;   // global int4
typedef __attribute__((address_space(3))) v4i lv4i_t;   // LDS int4

__device__ __forceinline__ unsigned short f2bf(float f) {
  unsigned int u = __float_as_uint(f);
  u += 0x7FFFu + ((u >> 16) & 1u);          // round-to-nearest-even
  return (unsigned short)(u >> 16);
}
__device__ __forceinline__ float bf2f(unsigned short h) {
  return __uint_as_float(((unsigned int)h) << 16);
}

// async global -> LDS 16B copy (CDNA5 GLOBAL_LOAD_ASYNC_TO_LDS_B128), with
// synchronous fallback if the builtin is unavailable on this toolchain.
__device__ __forceinline__ void g2l_b128(const unsigned short* gp, unsigned short* lp) {
#if ASYNC_LDS
  __builtin_amdgcn_global_load_async_to_lds_b128(
      (gv4i_t*)(unsigned long long)(uintptr_t)gp,
      (lv4i_t*)(unsigned int)(uintptr_t)lp, 0, 0);
#else
  *(u16x8*)lp = *(const u16x8*)gp;
#endif
}
__device__ __forceinline__ void wait_async() {
#if ASYNC_LDS
#  if __has_builtin(__builtin_amdgcn_s_wait_asynccnt)
  __builtin_amdgcn_s_wait_asynccnt(0);
#  else
  asm volatile("s_wait_asynccnt 0x0" ::: "memory");
#  endif
#endif
}

// ---------------------------------------------------------------------------
// H (f32, N x E) -> bf16 row-major + col-major via LDS-tiled 64x64 transpose.
// All global reads/writes fully coalesced.
#define TT 64
#define TPAD 68   // row stride (136B: 8B-aligned, 34-bank skew)
__global__ __launch_bounds__(256) void cvt_H_tiled(const float* __restrict__ Hg,
                                                   unsigned short* __restrict__ Hbf,
                                                   unsigned short* __restrict__ HbfT) {
  __shared__ unsigned short tile[TT * TPAD];
  const int n0 = blockIdx.x * TT;
  const int e0 = blockIdx.y * TT;
  const int t  = threadIdx.x;
#pragma unroll
  for (int p = 0; p < 4; ++p) {
    int c  = p * 256 + t;        // 0..1023
    int r  = c >> 4;             // node row in tile
    int c4 = (c & 15) * 4;       // edge col in tile
    f32x4 v = *(const f32x4*)(Hg + (size_t)(n0 + r) * NE + e0 + c4);
    u16x4 o;
    o[0] = f2bf(v[0]); o[1] = f2bf(v[1]); o[2] = f2bf(v[2]); o[3] = f2bf(v[3]);
    *(u16x4*)(Hbf + (size_t)(n0 + r) * NE + e0 + c4) = o;
    *(u16x4*)&tile[r * TPAD + c4] = o;
  }
  __syncthreads();
#pragma unroll
  for (int p = 0; p < 4; ++p) {
    int c  = p * 256 + t;
    int er = c >> 4;             // edge row in tile
    int c4 = (c & 15) * 4;       // node col in tile
    u16x4 o;
#pragma unroll
    for (int i = 0; i < 4; ++i) o[i] = tile[(c4 + i) * TPAD + er];
    *(u16x4*)(HbfT + (size_t)(e0 + er) * NN + n0 + c4) = o;
  }
}

// Generic f32 -> bf16 (count multiple of 4).
__global__ __launch_bounds__(256) void cvt_f32_bf16(const float* __restrict__ in,
                                                    unsigned short* __restrict__ out) {
  size_t base = ((size_t)blockIdx.x * 256 + threadIdx.x) * 4;
  f32x4 v = *(const f32x4*)(in + base);
  u16x4 o;
  o[0] = f2bf(v[0]); o[1] = f2bf(v[1]); o[2] = f2bf(v[2]); o[3] = f2bf(v[3]);
  *(u16x4*)(out + base) = o;
}

// deg_e[e] = sum_n H[n,e]  (read bf16 H^T row e, length NN)
__global__ __launch_bounds__(256) void col_sum(const unsigned short* __restrict__ HbfT,
                                               float* __restrict__ deg_e) {
  int e = blockIdx.x;
  const unsigned short* row = HbfT + (size_t)e * NN;
  float s = 0.f;
  for (int i = threadIdx.x * 8; i < NN; i += 256 * 8) {
    u16x8 v = *(const u16x8*)(row + i);
#pragma unroll
    for (int j = 0; j < 8; ++j) s += bf2f(v[j]);
  }
  __shared__ float sm[256];
  sm[threadIdx.x] = s;
  __syncthreads();
  for (int st = 128; st > 0; st >>= 1) {
    if (threadIdx.x < st) sm[threadIdx.x] += sm[threadIdx.x + st];
    __syncthreads();
  }
  if (threadIdx.x == 0) deg_e[e] = sm[0];
}

// Per edge: s = sigmoid(edge_emb[e,:] . V); hwvec=s; wde=s*deg_e; w_out=s
__global__ __launch_bounds__(128) void edge_gate(const float* __restrict__ edge_emb,
                                                 const float* __restrict__ V,
                                                 const float* __restrict__ deg_e,
                                                 float* __restrict__ hwvec,
                                                 float* __restrict__ wde,
                                                 float* __restrict__ w_out) {
  int e = blockIdx.x;
  int t = threadIdx.x;
  const float* row = edge_emb + (size_t)e * FT;
  f32x4 a = *(const f32x4*)(row + t * 4);
  f32x4 b = *(const f32x4*)(V + t * 4);
  float s = a[0] * b[0] + a[1] * b[1] + a[2] * b[2] + a[3] * b[3];
  __shared__ float sm[128];
  sm[t] = s;
  __syncthreads();
  for (int st = 64; st > 0; st >>= 1) {
    if (t < st) sm[t] += sm[t + st];
    __syncthreads();
  }
  if (t == 0) {
    float sig = 1.f / (1.f + __expf(-sm[0]));
    hwvec[e] = sig;
    wde[e]   = sig * deg_e[e];
    w_out[e] = sig;
  }
}

// deg_v[n] = Hbf[n,:] . hwvec
__global__ __launch_bounds__(256) void node_deg(const unsigned short* __restrict__ Hbf,
                                                const float* __restrict__ hwvec,
                                                float* __restrict__ deg_v) {
  int n = blockIdx.x;
  const unsigned short* row = Hbf + (size_t)n * NE;
  int i0 = threadIdx.x * 16;
  float s = 0.f;
  u16x8 a = *(const u16x8*)(row + i0);
  u16x8 b = *(const u16x8*)(row + i0 + 8);
#pragma unroll
  for (int j = 0; j < 8; ++j) s += bf2f(a[j]) * hwvec[i0 + j];
#pragma unroll
  for (int j = 0; j < 8; ++j) s += bf2f(b[j]) * hwvec[i0 + 8 + j];
  __shared__ float sm[256];
  sm[threadIdx.x] = s;
  __syncthreads();
  for (int st = 128; st > 0; st >>= 1) {
    if (threadIdx.x < st) sm[threadIdx.x] += sm[threadIdx.x + st];
    __syncthreads();
  }
  if (threadIdx.x == 0) deg_v[n] = sm[0];
}

// ---------------------------------------------------------------------------
// Tiled bf16 WMMA GEMM: C(MxNcols,f32 acc) = A(MxK,bf16 rm) * B(KxNcols,bf16 rm)
// Epilogue: optional per-row scale, optional per-col bias, f32 or bf16 store.
// Block tile 128x128, 8 waves, each wave 32x64 via 2x4 wmma_f32_16x16x32_bf16.
__global__ __launch_bounds__(256) void gemm_bf16(
    const unsigned short* __restrict__ A,
    const unsigned short* __restrict__ B,
    float* __restrict__ Cf,
    unsigned short* __restrict__ Cb,
    const float* __restrict__ rowScale,
    const float* __restrict__ bias,
    int M, int K, int Ncols, int storeBf16) {
  __shared__ unsigned short lA[TM * KPAD];  // [m][k]
  __shared__ unsigned short lB[TN * KPAD];  // B transposed: [n][k]

  const int tid  = threadIdx.x;
  const int row0 = blockIdx.x * TM;
  const int col0 = blockIdx.y * TN;
  const int wid  = tid >> 5;
  const int lane = tid & 31;
  const int lm   = lane & 15;
  const int h    = lane >> 4;
  const int waveM = wid & 3;   // 0..3 -> 32-row strip
  const int waveN = wid >> 2;  // 0..1 -> 64-col strip

  v8f acc[2][4];
#pragma unroll
  for (int s = 0; s < 2; ++s)
#pragma unroll
    for (int j = 0; j < 4; ++j) acc[s][j] = v8f{0.f, 0.f, 0.f, 0.f, 0.f, 0.f, 0.f, 0.f};

  union Frag { u16x8 u[2]; v16bf v; };

  for (int k0 = 0; k0 < K; k0 += TK) {
    // --- stage A tile: 128 rows x 32 bf16; 2 threads per row (async -> LDS) ---
    {
      int r = tid >> 1, half = tid & 1;
      const unsigned short* src = A + (size_t)(row0 + r) * K + k0 + half * 16;
      unsigned short* dst = &lA[r * KPAD + half * 16];
      g2l_b128(src, dst);
      g2l_b128(src + 8, dst + 8);
    }
    // --- stage B tile transposed: read (k, n..n+7) coalesced, scatter to [n][k] ---
#pragma unroll
    for (int c = tid; c < 512; c += 256) {
      int kk  = c >> 4;         // 0..31
      int nch = (c & 15) * 8;   // 0..120
      u16x8 v = *(const u16x8*)(B + (size_t)(k0 + kk) * Ncols + col0 + nch);
#pragma unroll
      for (int j = 0; j < 8; ++j) lB[(nch + j) * KPAD + kk] = v[j];
    }
    wait_async();
    __syncthreads();

    // --- fragments per ISA 16-bit A/B layouts ---
    Frag fa[2], fb[4];
#pragma unroll
    for (int s = 0; s < 2; ++s) {
      int r = waveM * 32 + s * 16 + lm;           // M row
      fa[s].u[0] = *(const u16x8*)&lA[r * KPAD + 8 * h];        // K = 8h..8h+7
      fa[s].u[1] = *(const u16x8*)&lA[r * KPAD + 16 + 8 * h];   // K = 16+8h..
    }
#pragma unroll
    for (int j = 0; j < 4; ++j) {
      int n = waveN * 64 + j * 16 + lm;           // N col
      fb[j].u[0] = *(const u16x8*)&lB[n * KPAD + 8 * h];        // K = 8h..8h+7
      fb[j].u[1] = *(const u16x8*)&lB[n * KPAD + 16 + 8 * h];   // K = 16+8h..
    }
#pragma unroll
    for (int s = 0; s < 2; ++s)
#pragma unroll
      for (int j = 0; j < 4; ++j)
        acc[s][j] = __builtin_amdgcn_wmma_f32_16x16x32_bf16(
            false, fa[s].v, false, fb[j].v, (short)0, acc[s][j], false, false);
    __syncthreads();
  }

  // --- epilogue: C layout VGPR r, lane l -> M = r + 8h, N = lm ---
  const int rowBase = row0 + waveM * 32;
  const int colBase = col0 + waveN * 64;
#pragma unroll
  for (int s = 0; s < 2; ++s) {
#pragma unroll
    for (int j = 0; j < 4; ++j) {
      int col  = colBase + j * 16 + lm;
      float bb = bias ? bias[col] : 0.f;
#pragma unroll
      for (int r = 0; r < 8; ++r) {
        int row = rowBase + s * 16 + h * 8 + r;
        float v = acc[s][j][r];
        if (rowScale) v *= rowScale[row];
        v += bb;
        size_t o = (size_t)row * Ncols + col;
        if (storeBf16) Cb[o] = f2bf(v);
        else           Cf[o] = v;
      }
    }
  }
}

// ---------------------------------------------------------------------------
extern "C" void kernel_launch(void* const* d_in, const int* in_sizes, int n_in,
                              void* d_out, int out_size, void* d_ws, size_t ws_size,
                              hipStream_t stream) {
  const float* x    = (const float*)d_in[0];  // N x FT
  const float* Hg   = (const float*)d_in[1];  // N x E
  const float* Wt   = (const float*)d_in[2];  // FT x FT
  const float* V    = (const float*)d_in[3];  // FT x 1
  const float* bias = (const float*)d_in[4];  // FT

  float* out   = (float*)d_out;                 // N x FT
  float* w_out = out + (size_t)NN * FT;         // E

  char* wsp = (char*)d_ws;
  auto alloc = [&](size_t bytes) -> char* {
    char* p = wsp;
    wsp += (bytes + 255) & ~(size_t)255;
    return p;
  };
  unsigned short* Hbf      = (unsigned short*)alloc((size_t)NN * NE * 2);
  unsigned short* HbfT     = (unsigned short*)alloc((size_t)NN * NE * 2);
  unsigned short* xbf      = (unsigned short*)alloc((size_t)NN * FT * 2);
  unsigned short* wbf      = (unsigned short*)alloc((size_t)FT * FT * 2);
  float*          edge_emb = (float*)alloc((size_t)NE * FT * 4);
  float*          hwvec    = (float*)alloc((size_t)NE * 4);
  float*          deg_e    = (float*)alloc((size_t)NE * 4);
  float*          wde      = (float*)alloc((size_t)NE * 4);
  float*          deg_v    = (float*)alloc((size_t)NN * 4);
  unsigned short* sc_bf    = (unsigned short*)alloc((size_t)NN * FT * 2);  // deg_v * (x@W)
  unsigned short* t_bf     = (unsigned short*)alloc((size_t)NE * FT * 2);  // (w*deg_e) * t

  dim3 blk(256);

  // Precision conversions (H via coalesced LDS-tiled transpose)
  cvt_H_tiled<<<dim3(NN / TT, NE / TT), blk, 0, stream>>>(Hg, Hbf, HbfT);
  cvt_f32_bf16<<<(size_t)NN * FT / 4 / 256, blk, 0, stream>>>(x, xbf);
  cvt_f32_bf16<<<(size_t)FT * FT / 4 / 256, blk, 0, stream>>>(Wt, wbf);

  // deg_e[e] = sum_n H[n,e]
  col_sum<<<NE, blk, 0, stream>>>(HbfT, deg_e);

  // edge_emb = H^T @ x
  gemm_bf16<<<dim3(NE / TM, FT / TN), blk, 0, stream>>>(
      HbfT, xbf, edge_emb, nullptr, nullptr, nullptr, NE, NN, FT, 0);

  // w = sigmoid(edge_emb @ V); wde = w * deg_e
  edge_gate<<<NE, dim3(128), 0, stream>>>(edge_emb, V, deg_e, hwvec, wde, w_out);

  // deg_v = H @ w
  node_deg<<<NN, blk, 0, stream>>>(Hbf, hwvec, deg_v);

  // sc = bf16(deg_v[:,None] * (x @ W))
  gemm_bf16<<<dim3(NN / TM, FT / TN), blk, 0, stream>>>(
      xbf, wbf, nullptr, sc_bf, deg_v, nullptr, NN, FT, FT, 1);

  // t = bf16((w*deg_e)[:,None] * (H^T @ sc))
  gemm_bf16<<<dim3(NE / TM, FT / TN), blk, 0, stream>>>(
      HbfT, sc_bf, nullptr, t_bf, wde, nullptr, NE, NN, FT, 1);

  // out = deg_v[:,None] * (H @ t) + bias
  gemm_bf16<<<dim3(NN / TM, FT / TN), blk, 0, stream>>>(
      Hbf, t_bf, out, nullptr, deg_v, bias, NN, NE, FT, 0);
}
